// Head_41145786695883
// MI455X (gfx1250) — compile-verified, matrix operand order checked by reading, not run on hardware
//
#include <hip/hip_runtime.h>

// ---------------------------------------------------------------------------
// Single-head attention for MI455X (gfx1250, wave32, WMMA).
//   x:[8,4096,768] f32 -> q,k,v (bf16 in ws) -> flash attention -> out f32.
// Matmuls on v_wmma_f32_16x16x32_bf16 (f32 accum). K/V blocks staged into LDS
// with double-buffered GLOBAL_LOAD_ASYNC_TO_LDS_B128 (ASYNCcnt). Softmax in
// exp2 domain (scale*log2e folded into q).
// ---------------------------------------------------------------------------

typedef __attribute__((ext_vector_type(16))) __bf16 v16bf;
typedef __attribute__((ext_vector_type(8)))  float  v8f;

#define N_EMBD 768
#define HEAD   64
#define BATCH  8
#define SEQ    4096
#define MTOT   (BATCH * SEQ)          // 32768 rows total
// 768^-0.5 * log2(e): logits land directly in log2 domain.
#define QSCALE 0.052058875f

static __device__ __forceinline__ v8f wmma_bf16(v16bf a, v16bf b, v8f c) {
  // (neg_a, A, neg_b, B, c_mod, C, reuse_a, reuse_b)
  return __builtin_amdgcn_wmma_f32_16x16x32_bf16(false, a, false, b, (short)0, c,
                                                 false, false);
}

// f32 -> bf16 via native hardware convert (v_cvt_pk_bf16_f32).
static __device__ __forceinline__ unsigned short f2u(float f) {
  return __builtin_bit_cast(unsigned short, (__bf16)f);
}
static __device__ __forceinline__ __bf16 u2b(unsigned short u) {
  return __builtin_bit_cast(__bf16, u);
}
static __device__ __forceinline__ v8f zero8() {
  v8f z = {0.f, 0.f, 0.f, 0.f, 0.f, 0.f, 0.f, 0.f};
  return z;
}

// K-pair base offsets for the 16-bit A/B fragment layout (ISA 7.12.2):
// half 0 covers K {0..7,16..23}, half 1 covers K {8..15,24..31}, 2 per VGPR.
static __device__ __forceinline__ void kk_pattern(int hf, int* kks) {
#pragma unroll
  for (int i = 0; i < 8; ++i)
    kks[i] = (i < 4 ? 2 * i : 16 + 2 * (i - 4)) + 8 * hf;
}

// Load one 16-bit B fragment (32xK rows x 16 cols) from a row pointer whose
// row holds contiguous K-pairs; kks holds this lane's K-pair offsets.
static __device__ __forceinline__ v16bf load_frag(const unsigned short* p,
                                                  const int* kks) {
  v16bf b;
#pragma unroll
  for (int i = 0; i < 8; ++i) {
    unsigned int u = *reinterpret_cast<const unsigned int*>(p + kks[i]);
    b[2 * i] = u2b((unsigned short)(u & 0xffffu));
    b[2 * i + 1] = u2b((unsigned short)(u >> 16));
  }
  return b;
}

// Low 32 bits of a flat pointer into LDS == LDS byte offset (aperture layout).
static __device__ __forceinline__ unsigned lds_addr(const void* p) {
  return (unsigned)(size_t)p;
}

// ---------------------------------------------------------------------------
// Kernel 0: W[768][64] f32 -> W^T[64][768] bf16 (contiguous K-pairs for B frags)
// ---------------------------------------------------------------------------
__global__ __launch_bounds__(256) void wcvt_kernel(
    const float* __restrict__ Wq, const float* __restrict__ Wk,
    const float* __restrict__ Wv, unsigned short* __restrict__ WqT,
    unsigned short* __restrict__ WkT, unsigned short* __restrict__ WvT) {
  int idx = blockIdx.x * 256 + threadIdx.x;
  if (idx >= 3 * N_EMBD * HEAD) return;
  int m = idx / (N_EMBD * HEAD);
  int r = idx % (N_EMBD * HEAD);
  int k = r / HEAD;
  int n = r % HEAD;
  const float* W = (m == 0) ? Wq : (m == 1 ? Wk : Wv);
  unsigned short* WT = (m == 0) ? WqT : (m == 1 ? WkT : WvT);
  WT[(size_t)n * N_EMBD + k] = f2u(W[(size_t)k * HEAD + n]);
}

// ---------------------------------------------------------------------------
// Kernel 1: fused QKV projection. One wave = one 16-row tile of [32768,768]x
// [768,64] for q,k,v simultaneously (x is read exactly once). Per k-step all
// 12 weight B-fragments are preloaded into distinct registers (one big load
// clause) so the 12 WMMAs issue back-to-back instead of load->wait->wmma.
// q scaled by QSCALE; v stored transposed [64][32768] for P@V B-fragments.
// ---------------------------------------------------------------------------
__global__ __launch_bounds__(128) void qkv_kernel(
    const float* __restrict__ x, const float* __restrict__ bq,
    const float* __restrict__ bk, const float* __restrict__ bv,
    const unsigned short* __restrict__ WqT, const unsigned short* __restrict__ WkT,
    const unsigned short* __restrict__ WvT, unsigned short* __restrict__ qo,
    unsigned short* __restrict__ ko, unsigned short* __restrict__ vT) {
  const int lane = threadIdx.x & 31;
  const int wid = threadIdx.x >> 5;
  const int row0 = (blockIdx.x * 4 + wid) * 16;
  const int m = lane & 15, hf = lane >> 4;
  int kks[8];
  kk_pattern(hf, kks);

  v8f acc[3][4];
#pragma unroll
  for (int w = 0; w < 3; ++w)
#pragma unroll
    for (int nt = 0; nt < 4; ++nt) acc[w][nt] = zero8();

  const float* xrow = x + (size_t)(row0 + m) * N_EMBD;
  for (int k0 = 0; k0 < N_EMBD; k0 += 32) {
    // A fragment: 16x32 slice of x, f32 -> bf16 (v_cvt_pk_bf16_f32).
    v16bf a;
#pragma unroll
    for (int i = 0; i < 8; ++i) {
      float2 f = *reinterpret_cast<const float2*>(xrow + k0 + kks[i]);
      a[2 * i] = (__bf16)f.x;
      a[2 * i + 1] = (__bf16)f.y;
    }
    // Preload all 12 B fragments (q,k,v x 4 n-tiles) into distinct registers.
    v16bf bfr[12];
#pragma unroll
    for (int w = 0; w < 3; ++w) {
      const unsigned short* WT = (w == 0) ? WqT : (w == 1 ? WkT : WvT);
#pragma unroll
      for (int nt = 0; nt < 4; ++nt)
        bfr[w * 4 + nt] = load_frag(WT + (size_t)(nt * 16 + m) * N_EMBD + k0, kks);
    }
    // 12 back-to-back WMMAs; next k-step's loads overlap with the matrix pipe.
#pragma unroll
    for (int w = 0; w < 3; ++w)
#pragma unroll
      for (int nt = 0; nt < 4; ++nt)
        acc[w][nt] = wmma_bf16(a, bfr[w * 4 + nt], acc[w][nt]);
  }

  // C/D layout: VGPR r = rows r (lanes 0-15) / r+8 (lanes 16-31), col = lane%16.
#pragma unroll
  for (int nt = 0; nt < 4; ++nt) {
    int col = nt * 16 + m;
    float biasq = bq[col], biask = bk[col], biasv = bv[col];
#pragma unroll
    for (int r = 0; r < 8; ++r) {
      int row = row0 + r + 8 * hf;
      qo[(size_t)row * HEAD + col] = f2u((acc[0][nt][r] + biasq) * QSCALE);
      ko[(size_t)row * HEAD + col] = f2u(acc[1][nt][r] + biask);
      vT[(size_t)col * MTOT + row] = f2u(acc[2][nt][r] + biasv);
    }
  }
}

// ---------------------------------------------------------------------------
// Kernel 2: flash attention. 4 waves/block, wave = 16 q rows, 32 keys/iter.
// K/V blocks staged in LDS via double-buffered async DMA; S via 4 WMMAs,
// online softmax in log2 domain (16-lane shfl_xor row reductions), P converted
// C/D->A layout through per-wave LDS, O += P@V.
// ---------------------------------------------------------------------------
__global__ __launch_bounds__(128) void attn_kernel(
    const unsigned short* __restrict__ q, const unsigned short* __restrict__ kg,
    const unsigned short* __restrict__ vg, float* __restrict__ out) {
  // Padded rows so the 16-lane fragment reads hit distinct LDS banks.
  __shared__ unsigned short kbuf[2][32][72];  // [buf][key][h]   (64 used + pad)
  __shared__ unsigned short vbuf[2][64][40];  // [buf][h][key]   (32 used + pad)
  __shared__ unsigned short pbuf[4][16][40];  // per-wave P transpose tile
  const int tid = threadIdx.x;
  const int lane = tid & 31;
  const int wid = tid >> 5;
  const int m = lane & 15, hf = lane >> 4;
  const int base = blockIdx.y * SEQ;
  const int q0 = blockIdx.x * 64 + wid * 16;
  int kks[8];
  kk_pattern(hf, kks);

  // Async-stage one 32-key block (K: 32x64 bf16, V^T: 64x32 bf16) into LDS.
  // 128 threads x (2 K + 2 V) GLOBAL_LOAD_ASYNC_TO_LDS_B128 per block.
  auto issue_block = [&](int kb, int buf) {
    const unsigned short* ksrc = kg + (size_t)(base + kb) * HEAD;
    const unsigned short* vsrc = vg + (size_t)(base + kb);
#pragma unroll
    for (int j = 0; j < 2; ++j) {
      int c = tid * 2 + j;             // 256 chunks of 16B
      int key = c >> 3, part = c & 7;  // 8 chunks per 64-half K row
      unsigned ldsa = lds_addr(&kbuf[buf][key][part * 8]);
      int voff = c * 16;               // K block is contiguous in global
      asm volatile("global_load_async_to_lds_b128 %0, %1, %2"
                   :: "v"(ldsa), "v"(voff), "s"(ksrc) : "memory");
    }
#pragma unroll
    for (int j = 0; j < 2; ++j) {
      int c = tid * 2 + j;
      int h = c >> 2, part = c & 3;    // 4 chunks per 32-half V^T row
      unsigned ldsa = lds_addr(&vbuf[buf][h][part * 8]);
      int voff = (h * MTOT + part * 8) * 2;
      asm volatile("global_load_async_to_lds_b128 %0, %1, %2"
                   :: "v"(ldsa), "v"(voff), "s"(vsrc) : "memory");
    }
  };

  // Q A-fragments (K = head dim 64 -> two 16x32 fragments), held for all keys.
  v16bf aq[2];
#pragma unroll
  for (int ks = 0; ks < 2; ++ks)
    aq[ks] = load_frag(q + (size_t)(base + q0 + m) * HEAD + ks * 32, kks);

  float mrow[8], lrow[8];
  v8f o[4];
#pragma unroll
  for (int r = 0; r < 8; ++r) { mrow[r] = -3.0e38f; lrow[r] = 0.f; }
#pragma unroll
  for (int nt = 0; nt < 4; ++nt) o[nt] = zero8();

  issue_block(0, 0);
  for (int kb = 0; kb < SEQ; kb += 32) {
    const int cur = (kb >> 5) & 1;
    if (kb + 32 < SEQ) {
      issue_block(kb + 32, cur ^ 1);   // overlap next copy with this compute
      asm volatile("s_wait_asynccnt 0x4" ::: "memory");  // prev 4 done
    } else {
      asm volatile("s_wait_asynccnt 0x0" ::: "memory");
    }
    __syncthreads();  // staged block visible to all 4 waves

    // S = Q * K^T for 32 keys: 2 n-tiles x 2 k-steps, fragments from LDS.
    v8f s[2];
    s[0] = zero8(); s[1] = zero8();
#pragma unroll
    for (int nt = 0; nt < 2; ++nt) {
      const unsigned short* kp = &kbuf[cur][nt * 16 + m][0];
#pragma unroll
      for (int ks = 0; ks < 2; ++ks)
        s[nt] = wmma_bf16(aq[ks], load_frag(kp + ks * 32, kks), s[nt]);
    }
    // Online softmax (exp2 domain; scale folded into q).
#pragma unroll
    for (int r = 0; r < 8; ++r) {
      float mx = fmaxf(s[0][r], s[1][r]);
#pragma unroll
      for (int off = 1; off < 16; off <<= 1)
        mx = fmaxf(mx, __shfl_xor(mx, off, 16));
      float mn = fmaxf(mrow[r], mx);
      float alpha = exp2f(mrow[r] - mn);
      float p0 = exp2f(s[0][r] - mn);
      float p1 = exp2f(s[1][r] - mn);
      s[0][r] = p0;
      s[1][r] = p1;
      float rs = p0 + p1;
#pragma unroll
      for (int off = 1; off < 16; off <<= 1) rs += __shfl_xor(rs, off, 16);
      lrow[r] = lrow[r] * alpha + rs;
      mrow[r] = mn;
#pragma unroll
      for (int nt = 0; nt < 4; ++nt) o[nt][r] *= alpha;
    }
    // P: C/D layout (f32) -> per-wave LDS tile -> A-fragment layout (bf16).
#pragma unroll
    for (int nt = 0; nt < 2; ++nt)
#pragma unroll
      for (int r = 0; r < 8; ++r)
        pbuf[wid][r + 8 * hf][nt * 16 + m] = f2u(s[nt][r]);
    __builtin_amdgcn_wave_barrier();
    asm volatile("s_wait_dscnt 0" ::: "memory");
    v16bf ap;
#pragma unroll
    for (int i = 0; i < 8; ++i) {
      ap[2 * i] = u2b(pbuf[wid][m][kks[i]]);
      ap[2 * i + 1] = u2b(pbuf[wid][m][kks[i] + 1]);
    }
    // O += P * V : V^T rows in LDS give contiguous key-pairs per B fragment.
#pragma unroll
    for (int nt = 0; nt < 4; ++nt)
      o[nt] = wmma_bf16(ap, load_frag(&vbuf[cur][nt * 16 + m][0], kks), o[nt]);
    __syncthreads();  // all waves done with buf `cur` before it is refilled
  }
  // Normalize and store f32 output.
#pragma unroll
  for (int r = 0; r < 8; ++r) {
    float inv = 1.0f / lrow[r];
    int row = base + q0 + r + 8 * hf;
#pragma unroll
    for (int nt = 0; nt < 4; ++nt)
      out[(size_t)row * HEAD + nt * 16 + m] = o[nt][r] * inv;
  }
}

// ---------------------------------------------------------------------------
extern "C" void kernel_launch(void* const* d_in, const int* in_sizes, int n_in,
                              void* d_out, int out_size, void* d_ws,
                              size_t ws_size, hipStream_t stream) {
  (void)in_sizes; (void)n_in; (void)out_size; (void)ws_size;
  const float* x  = (const float*)d_in[0];
  const float* Wq = (const float*)d_in[1];
  const float* bq = (const float*)d_in[2];
  const float* Wk = (const float*)d_in[3];
  const float* bk = (const float*)d_in[4];
  const float* Wv = (const float*)d_in[5];
  const float* bv = (const float*)d_in[6];
  float* out = (float*)d_out;

  // Workspace: q,k bf16 [32768][64]; vT bf16 [64][32768]; 3x W^T bf16 [64][768].
  char* ws = (char*)d_ws;
  unsigned short* qo  = (unsigned short*)(ws);
  unsigned short* ko  = (unsigned short*)(ws + (size_t)MTOT * HEAD * 2);
  unsigned short* vT  = (unsigned short*)(ws + (size_t)MTOT * HEAD * 4);
  unsigned short* WqT = (unsigned short*)(ws + (size_t)MTOT * HEAD * 6);
  unsigned short* WkT = WqT + (size_t)N_EMBD * HEAD;
  unsigned short* WvT = WkT + (size_t)N_EMBD * HEAD;

  hipLaunchKernelGGL(wcvt_kernel, dim3((3 * N_EMBD * HEAD + 255) / 256),
                     dim3(256), 0, stream, Wq, Wk, Wv, WqT, WkT, WvT);
  hipLaunchKernelGGL(qkv_kernel, dim3(MTOT / 64), dim3(128), 0, stream, x, bq,
                     bk, bv, WqT, WkT, WvT, qo, ko, vT);
  hipLaunchKernelGGL(attn_kernel, dim3(SEQ / 64, BATCH), dim3(128), 0, stream,
                     qo, ko, vT, out);
}
